// MultiHeadSelfAttention_75307956568524
// MI455X (gfx1250) — matrix operand order repaired and checked by reading, not compiled, not run on hardware
//
#include <hip/hip_runtime.h>
#include <math.h>

typedef __bf16 bf16_t;
typedef __attribute__((ext_vector_type(16))) __bf16 v16bf;
typedef __attribute__((ext_vector_type(8)))  __bf16 v8bf;
typedef __attribute__((ext_vector_type(8)))  float  v8f;

#define BZ 2
#define SEQ 2048
#define EMB 768
#define NH 12
#define HDIM 64
#define TOK (BZ*SEQ)        // 4096
#define E3  (3*EMB)         // 2304

// ---------------- fragment helpers (wave32 WMMA bf16 layouts) ----------------

// A-frag 16x32: caller passes pointer to (row, k0 + 8*laneHalf); elements
// 0..7 = K chunk at p, 8..15 = K chunk at p+16.
__device__ inline v16bf load_a_frag(const bf16_t* p) {
    v8bf lo = *(const v8bf*)p;
    v8bf hi = *(const v8bf*)(p + 16);
    return __builtin_shufflevector(lo, hi, 0,1,2,3,4,5,6,7,8,9,10,11,12,13,14,15);
}

// B-frag 32x16: lane holds one N column; caller passes (col, k0 + 8*laneHalf);
// elements 0..7 = K chunk at p, 8..15 = K chunk at p+16.
__device__ inline v16bf load_b_frag(const bf16_t* p) {
    v8bf lo = *(const v8bf*)p;
    v8bf hi = *(const v8bf*)(p + 16);
    return __builtin_shufflevector(lo, hi, 0,1,2,3,4,5,6,7,8,9,10,11,12,13,14,15);
}

__device__ inline v8f wmma_bf16(v16bf a, v16bf b, v8f c) {
    return __builtin_amdgcn_wmma_f32_16x16x32_bf16(false, a, false, b, (short)0, c, false, false);
}

__device__ inline float half_max(float v) {   // max over 16-lane half
    v = fmaxf(v, __shfl_xor(v, 1));
    v = fmaxf(v, __shfl_xor(v, 2));
    v = fmaxf(v, __shfl_xor(v, 4));
    v = fmaxf(v, __shfl_xor(v, 8));
    return v;
}
__device__ inline float half_sum(float v) {   // sum over 16-lane half
    v += __shfl_xor(v, 1);
    v += __shfl_xor(v, 2);
    v += __shfl_xor(v, 4);
    v += __shfl_xor(v, 8);
    return v;
}

// ---------------- kernel 1: LayerNorm -> bf16 ----------------

__global__ void ln_kernel(const float* __restrict__ x,
                          const float* __restrict__ gamma,
                          const float* __restrict__ beta,
                          bf16_t* __restrict__ xn) {
    int row = blockIdx.x;                 // token
    const float* xr = x + (size_t)row * EMB;
    float vals[3];
    float s = 0.f, s2 = 0.f;
    #pragma unroll
    for (int i = 0; i < 3; i++) {
        float v = xr[threadIdx.x + i * 256];
        vals[i] = v; s += v; s2 += v * v;
    }
    #pragma unroll
    for (int m = 1; m < 32; m <<= 1) { s += __shfl_xor(s, m); s2 += __shfl_xor(s2, m); }
    __shared__ float ws1[8], ws2[8];
    int w = threadIdx.x >> 5, l = threadIdx.x & 31;
    if (l == 0) { ws1[w] = s; ws2[w] = s2; }
    __syncthreads();
    if (w == 0) {
        float a = (l < 8) ? ws1[l] : 0.f;
        float b = (l < 8) ? ws2[l] : 0.f;
        #pragma unroll
        for (int m = 1; m < 8; m <<= 1) { a += __shfl_xor(a, m); b += __shfl_xor(b, m); }
        if (l == 0) { ws1[0] = a; ws2[0] = b; }
    }
    __syncthreads();
    float mu  = ws1[0] * (1.0f / EMB);
    float var = ws2[0] * (1.0f / EMB) - mu * mu;
    float rs  = rsqrtf(var + 1e-5f);
    #pragma unroll
    for (int i = 0; i < 3; i++) {
        int c = threadIdx.x + i * 256;
        float v = (vals[i] - mu) * rs * gamma[c] + beta[c];
        xn[(size_t)row * EMB + c] = (bf16_t)v;
    }
}

// ---------------- kernel 2: fp32 -> bf16 weight conversion ----------------

__global__ void cvt_kernel(const float* __restrict__ in, bf16_t* __restrict__ out, int n) {
    int i = blockIdx.x * blockDim.x + threadIdx.x;
    if (i < n) out[i] = (bf16_t)in[i];
}

// ---------------- kernel 3: QKV GEMM (32x64/wave) + bias + scatter to q/k/vT ----
// C = xn[4096x768] @ Wqkv^T[768x2304] + bqkv, scatter bf16 into
// q,k: [B*H, S, HD]   vT: [B*H, HD, S]

__global__ void qkv_gemm(const bf16_t* __restrict__ A,  // [TOK, EMB]
                         const bf16_t* __restrict__ W,  // [E3, EMB] row-major
                         const float*  __restrict__ bias,
                         bf16_t* __restrict__ q, bf16_t* __restrict__ k,
                         bf16_t* __restrict__ vt) {
    int wave = threadIdx.x >> 5, lane = threadIdx.x & 31;
    int lm = lane & 15, lh = lane >> 4;
    const int TN = E3 / 64, TM = TOK / 32;          // 36 x 128 tiles
    int tile = blockIdx.x * 8 + wave;
    if (tile >= TM * TN) return;                    // wave-uniform
    int m0 = (tile / TN) * 32, n0 = (tile % TN) * 64;

    v8f c[2][4] = {};
    for (int kk = 0; kk < EMB; kk += 32) {
        v16bf a[2], b[4];
        #pragma unroll
        for (int i = 0; i < 2; i++)
            a[i] = load_a_frag(A + (size_t)(m0 + i * 16 + lm) * EMB + kk + 8 * lh);
        #pragma unroll
        for (int j = 0; j < 4; j++)
            b[j] = load_b_frag(W + (size_t)(n0 + j * 16 + lm) * EMB + kk + 8 * lh);
        #pragma unroll
        for (int i = 0; i < 2; i++)
            #pragma unroll
            for (int j = 0; j < 4; j++)
                c[i][j] = wmma_bf16(a[i], b[j], c[i][j]);
    }
    #pragma unroll
    for (int i = 0; i < 2; i++)
      #pragma unroll
      for (int j = 0; j < 4; j++)
        #pragma unroll
        for (int r = 0; r < 8; r++) {
            int m = m0 + i * 16 + r + 8 * lh;       // token
            int n = n0 + j * 16 + lm;               // qkv feature
            float val = c[i][j][r] + bias[n];
            int h = n / (3 * HDIM), rem = n % (3 * HDIM);
            int which = rem / HDIM, d = rem % HDIM;
            int b2 = m / SEQ, sidx = m % SEQ;
            int bh = b2 * NH + h;
            bf16_t bv = (bf16_t)val;
            if      (which == 0) q [((size_t)bh * SEQ + sidx) * HDIM + d] = bv;
            else if (which == 1) k [((size_t)bh * SEQ + sidx) * HDIM + d] = bv;
            else                 vt[((size_t)bh * HDIM + d) * SEQ + sidx] = bv;
        }
}

// ---------------- kernel 4: flash attention (per-wave 16 query rows) ----------------

__global__ void attn_kernel(const bf16_t* __restrict__ q,
                            const bf16_t* __restrict__ k,
                            const bf16_t* __restrict__ vt,
                            bf16_t* __restrict__ ctx) {   // [TOK, EMB]
    __shared__ bf16_t plds[8 * 512];                      // 16x32 probs per wave
    int wave = threadIdx.x >> 5, lane = threadIdx.x & 31;
    int lm = lane & 15, lh = lane >> 4;
    const int blocksPerBH = SEQ / 128;                    // 16
    int bh = blockIdx.x / blocksPerBH;
    int q0 = (blockIdx.x % blocksPerBH) * 128 + wave * 16;
    const bf16_t* qb = q  + (size_t)bh * SEQ * HDIM;
    const bf16_t* kb = k  + (size_t)bh * SEQ * HDIM;
    const bf16_t* vb = vt + (size_t)bh * HDIM * SEQ;

    v16bf qa[2];
    #pragma unroll
    for (int dstep = 0; dstep < 2; dstep++)
        qa[dstep] = load_a_frag(qb + (size_t)(q0 + lm) * HDIM + dstep * 32 + 8 * lh);

    v8f o[4] = {};
    float mx[8], ls[8];
    #pragma unroll
    for (int r = 0; r < 8; r++) { mx[r] = -1e30f; ls[r] = 0.f; }
    bf16_t* myp = plds + wave * 512;
    const float SC = 0.125f;            // 1/sqrt(64)
    const float L2E = 1.44269504f;

    for (int t0 = 0; t0 < SEQ; t0 += 32) {
        v8f s0 = {}, s1 = {};
        #pragma unroll
        for (int dstep = 0; dstep < 2; dstep++) {
            v16bf kb0 = load_b_frag(kb + (size_t)(t0 + lm)      * HDIM + dstep * 32 + 8 * lh);
            v16bf kb1 = load_b_frag(kb + (size_t)(t0 + 16 + lm) * HDIM + dstep * 32 + 8 * lh);
            s0 = wmma_bf16(qa[dstep], kb0, s0);
            s1 = wmma_bf16(qa[dstep], kb1, s1);
        }
        #pragma unroll
        for (int r = 0; r < 8; r++) {
            float a = s0[r] * SC, b = s1[r] * SC;
            float mn = fmaxf(mx[r], half_max(fmaxf(a, b)));
            float f  = exp2f((mx[r] - mn) * L2E);
            float p0 = exp2f((a - mn) * L2E);
            float p1 = exp2f((b - mn) * L2E);
            ls[r] = ls[r] * f + half_sum(p0 + p1);
            mx[r] = mn;
            #pragma unroll
            for (int dt = 0; dt < 4; dt++) o[dt][r] *= f;
            int mrow = r + 8 * lh;
            myp[mrow * 32 + lm]      = (bf16_t)p0;
            myp[mrow * 32 + 16 + lm] = (bf16_t)p1;
        }
        v16bf pa = load_a_frag(myp + lm * 32 + 8 * lh);   // D-layout -> A-layout via LDS
        #pragma unroll
        for (int dt = 0; dt < 4; dt++) {
            v16bf vf = load_b_frag(vb + (size_t)(dt * 16 + lm) * SEQ + t0 + 8 * lh);
            o[dt] = wmma_bf16(pa, vf, o[dt]);
        }
    }
    int h = bh % NH, b = bh / NH;
    #pragma unroll
    for (int r = 0; r < 8; r++) {
        float inv = 1.0f / ls[r];
        int token = b * SEQ + q0 + r + 8 * lh;
        #pragma unroll
        for (int dt = 0; dt < 4; dt++)
            ctx[(size_t)token * EMB + h * HDIM + dt * 16 + lm] = (bf16_t)(o[dt][r] * inv);
    }
}

// ---------------- kernel 5: output projection (32x64/wave) -> fp32 out ----------------

__global__ void proj_gemm(const bf16_t* __restrict__ A,  // [TOK, EMB]
                          const bf16_t* __restrict__ W,  // [EMB, EMB]
                          const float*  __restrict__ bias,
                          float* __restrict__ out) {     // [TOK, EMB] fp32
    int wave = threadIdx.x >> 5, lane = threadIdx.x & 31;
    int lm = lane & 15, lh = lane >> 4;
    const int TN = EMB / 64, TM = TOK / 32;         // 12 x 128 tiles
    int tile = blockIdx.x * 8 + wave;
    if (tile >= TM * TN) return;
    int m0 = (tile / TN) * 32, n0 = (tile % TN) * 64;

    v8f c[2][4] = {};
    for (int kk = 0; kk < EMB; kk += 32) {
        v16bf a[2], b[4];
        #pragma unroll
        for (int i = 0; i < 2; i++)
            a[i] = load_a_frag(A + (size_t)(m0 + i * 16 + lm) * EMB + kk + 8 * lh);
        #pragma unroll
        for (int j = 0; j < 4; j++)
            b[j] = load_b_frag(W + (size_t)(n0 + j * 16 + lm) * EMB + kk + 8 * lh);
        #pragma unroll
        for (int i = 0; i < 2; i++)
            #pragma unroll
            for (int j = 0; j < 4; j++)
                c[i][j] = wmma_bf16(a[i], b[j], c[i][j]);
    }
    #pragma unroll
    for (int i = 0; i < 2; i++)
      #pragma unroll
      for (int j = 0; j < 4; j++)
        #pragma unroll
        for (int r = 0; r < 8; r++) {
            int m = m0 + i * 16 + r + 8 * lh;
            int n = n0 + j * 16 + lm;
            out[(size_t)m * EMB + n] = c[i][j][r] + bias[n];
        }
}

// ---------------- host launcher ----------------

extern "C" void kernel_launch(void* const* d_in, const int* in_sizes, int n_in,
                              void* d_out, int out_size, void* d_ws, size_t ws_size,
                              hipStream_t stream) {
    const float* x     = (const float*)d_in[0];
    const float* gamma = (const float*)d_in[1];
    const float* beta  = (const float*)d_in[2];
    const float* Wqkv  = (const float*)d_in[3];
    const float* bqkv  = (const float*)d_in[4];
    const float* Wo    = (const float*)d_in[5];
    const float* bo    = (const float*)d_in[6];
    float* out = (float*)d_out;

    char* ws = (char*)d_ws;
    size_t off = 0;
    auto alloc = [&](size_t bytes) {
        char* p = ws + off;
        off += (bytes + 255) & ~(size_t)255;
        return p;
    };
    bf16_t* xn     = (bf16_t*)alloc((size_t)TOK * EMB * 2);
    bf16_t* Wqkv_b = (bf16_t*)alloc((size_t)E3  * EMB * 2);
    bf16_t* Wo_b   = (bf16_t*)alloc((size_t)EMB * EMB * 2);
    bf16_t* qbuf   = (bf16_t*)alloc((size_t)BZ * NH * SEQ * HDIM * 2);
    bf16_t* kbuf   = (bf16_t*)alloc((size_t)BZ * NH * SEQ * HDIM * 2);
    bf16_t* vtbuf  = (bf16_t*)alloc((size_t)BZ * NH * HDIM * SEQ * 2);
    bf16_t* ctx    = (bf16_t*)alloc((size_t)TOK * EMB * 2);
    (void)ws_size; (void)in_sizes; (void)n_in; (void)out_size;

    ln_kernel<<<TOK, 256, 0, stream>>>(x, gamma, beta, xn);
    cvt_kernel<<<(E3 * EMB + 255) / 256, 256, 0, stream>>>(Wqkv, Wqkv_b, E3 * EMB);
    cvt_kernel<<<(EMB * EMB + 255) / 256, 256, 0, stream>>>(Wo, Wo_b, EMB * EMB);
    qkv_gemm<<<(TOK / 32) * (E3 / 64) / 8, 256, 0, stream>>>(xn, Wqkv_b, bqkv, qbuf, kbuf, vtbuf);
    attn_kernel<<<(BZ * NH) * (SEQ / 128), 256, 0, stream>>>(qbuf, kbuf, vtbuf, ctx);
    proj_gemm<<<(TOK / 32) * (EMB / 64) / 8, 256, 0, stream>>>(ctx, Wo_b, bo, out);
}
